// TTTWrapper_36086315221744
// MI455X (gfx1250) — matrix operand order repaired and checked by reading, not compiled, not run on hardware
//
#include <hip/hip_runtime.h>
#include <hip/hip_bf16.h>

// ===================== Problem constants =====================
#define HNUM   16
#define HDIM   64
#define CHUNK  64
#define NCHUNK 128
#define LSEQ   8192
#define DIM    1024
#define BATCH  4
#define NTOK   32768L   // BATCH * LSEQ

// ===================== WMMA types & helpers (gfx1250, wave32) =====================
typedef __attribute__((ext_vector_type(16))) __bf16 v16bf;
typedef __attribute__((ext_vector_type(8)))  float  v8f;

union FragBF { v16bf v; unsigned int u[8]; };

__device__ __forceinline__ unsigned int pack_bf16(float lo, float hi) {
  unsigned int a = __float_as_uint(lo);
  unsigned int b = __float_as_uint(hi);
  a = (a + 0x7FFFu + ((a >> 16) & 1u)) >> 16;   // RNE
  b = (b + 0x7FFFu + ((b >> 16) & 1u)) >> 16;
  return a | (b << 16);
}

__device__ __forceinline__ v8f wmma_bf16(v16bf a, v16bf b, v8f c) {
  // (neg_a, A, neg_b, B, c_mod, C, reuse_a, reuse_b)
  return __builtin_amdgcn_wmma_f32_16x16x32_bf16(false, a, false, b, (short)0, c, false, false);
}

// ---- CDNA5 async copy: global -> LDS, tracked by ASYNCcnt ----
__device__ __forceinline__ void async_load_b128(unsigned lds_addr, const void* gptr) {
  asm volatile("global_load_async_to_lds_b128 %0, %1, off"
               :: "v"(lds_addr), "v"(gptr) : "memory");
}
__device__ __forceinline__ void wait_async0() {
  asm volatile("s_wait_asynccnt 0x0" ::: "memory");
}

// ---- fragment gathers from fp32 LDS, on-the-fly bf16 pack (used by scan kernel) ----
// A matrix 16x32 layout: lane = h*16+m ; VGPR p holds K = h*16 + 2p (+1)
__device__ __forceinline__ v16bf load_A_f32(const float* src, int ld, int mBase, int kBase) {
  int lane = threadIdx.x & 31;
  int h = lane >> 4, m = mBase + (lane & 15);
  FragBF f;
#pragma unroll
  for (int p = 0; p < 8; ++p) {
    int k = kBase + h * 16 + (p << 1);
    f.u[p] = pack_bf16(src[m * ld + k], src[m * ld + k + 1]);
  }
  return f.v;
}
// B matrix 32x16: lane = h*16+n ; VGPR p holds K = h*16 + 2p (+1). src is row-major B[k][n].
__device__ __forceinline__ v16bf load_B_f32(const float* src, int ld, int nBase, int kBase) {
  int lane = threadIdx.x & 31;
  int h = lane >> 4, n = nBase + (lane & 15);
  FragBF f;
#pragma unroll
  for (int p = 0; p < 8; ++p) {
    int k = kBase + h * 16 + (p << 1);
    f.u[p] = pack_bf16(src[k * ld + n], src[(k + 1) * ld + n]);
  }
  return f.v;
}
// B fragment where we have B^T row-major: src[n][k]
__device__ __forceinline__ v16bf load_Bt_f32(const float* src, int ld, int nBase, int kBase) {
  int lane = threadIdx.x & 31;
  int h = lane >> 4, n = nBase + (lane & 15);
  FragBF f;
#pragma unroll
  for (int p = 0; p < 8; ++p) {
    int k = kBase + h * 16 + (p << 1);
    f.u[p] = pack_bf16(src[n * ld + k], src[n * ld + k + 1]);
  }
  return f.v;
}
// A = (e .* xk)^T : A[m][k] = e[k] * xk[k][m]   (xk row-major 64x64)
__device__ __forceinline__ v16bf load_A_xkeT(const float* xk, const float* e, int mBase, int kBase) {
  int lane = threadIdx.x & 31;
  int h = lane >> 4, m = mBase + (lane & 15);
  FragBF f;
#pragma unroll
  for (int p = 0; p < 8; ++p) {
    int k = kBase + h * 16 + (p << 1);
    float lo = e[k]     * xk[k * 64 + m];
    float hi = e[k + 1] * xk[(k + 1) * 64 + m];
    f.u[p] = pack_bf16(lo, hi);
  }
  return f.v;
}
// D tile store: lane = h*16+n ; VGPR v -> row v+8h, col n (within 16x16)
__device__ __forceinline__ void store_tile64(float* dst, const v8f* acc, int wave, int lane,
                                             const float* addb) {
  int h = lane >> 4;
  int r0 = wave * 16 + 8 * h;
  int c0 = lane & 15;
#pragma unroll
  for (int t = 0; t < 4; ++t) {
    int c = c0 + t * 16;
    float bb = addb ? addb[c] : 0.f;
#pragma unroll
    for (int v = 0; v < 8; ++v) dst[(r0 + v) * 64 + c] = acc[t][v] + bb;
  }
}

// ===================== K0: f32 -> bf16 convert =====================
__global__ void cvt_f32_bf16(const float* __restrict__ src, unsigned short* __restrict__ dst, long n) {
  long i = (long)blockIdx.x * blockDim.x + threadIdx.x;
  long stride = (long)gridDim.x * blockDim.x;
  for (; i < n; i += stride) {
    unsigned int u = __float_as_uint(src[i]);
    u = (u + 0x7FFFu + ((u >> 16) & 1u)) >> 16;
    dst[i] = (unsigned short)u;
  }
}

// ===================== K1: bf16 WMMA GEMM, 128x128 tile, async double-buffered LDS ==========
// out[M,N] = A[M,K] @ Bt[N,K]^T + bias ; 256 threads = 8 waves (2x4), each wave 4x2 16x16 tiles.
// LDS (dynamic, 64 KB): buf s: As 128x64 bf16 (16KB) + Bs 128x64 bf16 (16KB).
__device__ __forceinline__ void issue_stage(const unsigned short* Ab, const unsigned short* Bb,
                                            int m0, int n0, int K, int kk,
                                            unsigned ldsA, unsigned ldsB) {
  const int tid = threadIdx.x;
#pragma unroll
  for (int i = 0; i < 4; ++i) {
    int idx = tid + i * 256;             // 0..1023 : 128 rows x 8 x (8 bf16)
    int r = idx >> 3, c = idx & 7;
    const unsigned short* ga = Ab + (size_t)(m0 + r) * K + kk + c * 8;
    const unsigned short* gb = Bb + (size_t)(n0 + r) * K + kk + c * 8;
    unsigned loff = (unsigned)(r * 128 + c * 16);   // bytes within tile
    async_load_b128(ldsA + loff, ga);
    async_load_b128(ldsB + loff, gb);
  }
}

__global__ __launch_bounds__(256) void gemm_bf16_wmma(
    const unsigned short* __restrict__ A,   // M x K bf16 row-major
    const unsigned short* __restrict__ Bt,  // N x K bf16 row-major (i.e. B^T)
    const float* __restrict__ bias,         // N
    float* __restrict__ out,                // M x N f32
    int M, int N, int K)
{
  extern __shared__ unsigned int smemU[];   // 16384 u32 = 64 KB (2 x (As+Bs))
  const int tid = threadIdx.x;
  const int wave = tid >> 5, lane = tid & 31;
  const int wm = wave >> 2, wn = wave & 3;   // 2 x 4 wave grid
  const int h = lane >> 4;
  const int m0 = blockIdx.y * 128;
  const int n0 = blockIdx.x * 128;
  const unsigned ldsBase = (unsigned)(unsigned long long)(void*)smemU;

  v8f acc[4][2] = {};

  issue_stage(A, Bt, m0, n0, K, 0, ldsBase, ldsBase + 16384u);

  const int S = K >> 6;   // 64-wide K stages
  for (int s = 0; s < S; ++s) {
    const unsigned int* Acur = smemU + (s & 1) * 8192;
    const unsigned int* Bcur = Acur + 4096;
    wait_async0();        // my stage-s async loads have landed in LDS
    __syncthreads();      // everyone's landed; everyone done computing stage s-1
    if (s + 1 < S) {
      unsigned boff = (unsigned)(((s + 1) & 1) * 32768);
      issue_stage(A, Bt, m0, n0, K, (s + 1) << 6, ldsBase + boff, ldsBase + boff + 16384u);
    }
#pragma unroll
    for (int ks = 0; ks < 2; ++ks) {
      const int kb = ks * 32;
      FragBF fa[4], fb[2];
#pragma unroll
      for (int mt = 0; mt < 4; ++mt) {
        int ml = wm * 64 + mt * 16 + (lane & 15);
#pragma unroll
        for (int p = 0; p < 8; ++p) {
          int k = kb + h * 16 + (p << 1);
          fa[mt].u[p] = Acur[ml * 32 + (k >> 1)];
        }
      }
#pragma unroll
      for (int nt = 0; nt < 2; ++nt) {
        int nl = wn * 32 + nt * 16 + (lane & 15);
#pragma unroll
        for (int p = 0; p < 8; ++p) {
          int k = kb + h * 16 + (p << 1);
          fb[nt].u[p] = Bcur[nl * 32 + (k >> 1)];
        }
      }
#pragma unroll
      for (int mt = 0; mt < 4; ++mt)
#pragma unroll
        for (int nt = 0; nt < 2; ++nt)
          acc[mt][nt] = wmma_bf16(fa[mt].v, fb[nt].v, acc[mt][nt]);
    }
  }

#pragma unroll
  for (int mt = 0; mt < 4; ++mt) {
    int row = m0 + wm * 64 + mt * 16 + 8 * h;
#pragma unroll
    for (int nt = 0; nt < 2; ++nt) {
      int col = n0 + wn * 32 + nt * 16 + (lane & 15);
      float bv = bias[col];
#pragma unroll
      for (int v = 0; v < 8; ++v)
        out[(size_t)(row + v) * N + col] = acc[mt][nt][v] + bv;
    }
  }
}

// ===================== K2: eta = sigmoid(hs . lrw_h + lrb_h) / (HD*C) =====================
__global__ __launch_bounds__(256) void eta_kernel(const float* __restrict__ hs,
                                                  const float* __restrict__ lrw,
                                                  const float* __restrict__ lrb,
                                                  float* __restrict__ eta) {
  __shared__ float part[256];
  long tok = blockIdx.x;
  int tid = threadIdx.x;
  int hh = tid >> 4, j = tid & 15;
  const float* x = hs + tok * DIM;
  const float* w = lrw + hh * DIM;
  float s = 0.f;
  int base = j * 64;
#pragma unroll 4
  for (int k = 0; k < 64; ++k) s += x[base + k] * w[base + k];
  part[tid] = s;
  __syncthreads();
  if (j == 0) {
    float t = 0.f;
#pragma unroll
    for (int q = 0; q < 16; ++q) t += part[hh * 16 + q];
    t += lrb[hh];
    float sig = 1.f / (1.f + expf(-t));
    eta[(long)hh * NTOK + tok] = sig * (1.f / 4096.f);  // BASE_LR/HD/C
  }
}

// ===================== K3: l2norm(XQ,XK) + ttt-norm of XV (in place) =====================
__global__ __launch_bounds__(256) void prep_kernel(float* __restrict__ XQ, float* __restrict__ XK,
                                                   float* __restrict__ XV,
                                                   const float* __restrict__ tnw,
                                                   const float* __restrict__ tnb) {
  long idx = (long)blockIdx.x * 256 + threadIdx.x;   // (token, head) row
  if (idx >= NTOK * HNUM) return;
  long tok = idx >> 4;
  int hh = idx & 15;
  float* q = XQ + tok * DIM + hh * HDIM;
  float* k = XK + tok * DIM + hh * HDIM;
  float* v = XV + tok * DIM + hh * HDIM;
  const float* g  = tnw + hh * HDIM;
  const float* bb = tnb + hh * HDIM;
  float sq = 0.f, sk = 0.f;
  for (int d = 0; d < HDIM; ++d) { sq += q[d] * q[d]; sk += k[d] * k[d]; }
  float invq = 1.f / fmaxf(sqrtf(sq), 1e-12f);
  float invk = 1.f / fmaxf(sqrtf(sk), 1e-12f);
  float s1 = 0.f, s2 = 0.f;
  for (int d = 0; d < HDIM; ++d) {
    q[d] *= invq;
    float kn = k[d] * invk; k[d] = kn;
    float dv = v[d] - kn;
    s1 += dv; s2 += dv * dv;
  }
  float mu = s1 * (1.f / 64.f);
  float varu = fmaxf((s2 - s1 * mu) * (1.f / 63.f), 0.f);  // ddof=1
  float inv = 1.f / (sqrtf(varu) + 1e-5f);
  for (int d = 0; d < HDIM; ++d) {
    float dv = v[d] - k[d];
    v[d] = g[d] * ((dv - mu) * inv) + bb[d] + k[d];
  }
}

// ===================== K4: TTT scan, one block per (b,h), W state in LDS =====================
__global__ __launch_bounds__(128) void ttt_scan(
    const float* __restrict__ XQ, const float* __restrict__ XK, const float* __restrict__ XV,
    const float* __restrict__ eta,                 // [H][NTOK]
    const float* __restrict__ tnw, const float* __restrict__ tnb,  // [H][64]
    const float* __restrict__ W1,  const float* __restrict__ b1,   // [H][64][64], [H][64]
    float* __restrict__ Y)                          // [NTOK][DIM]
{
  extern __shared__ float smem[];
  float* Wf    = smem;               // 4096
  float* xqf   = Wf    + 4096;       // 4096
  float* xkf   = xqf   + 4096;       // 4096
  float* Z1f   = xkf   + 4096;       // 4096 (Z1, then Z1_bar)
  float* gradf = Z1f   + 4096;       // 4096
  float* M2f   = gradf + 4096;       // 4096 (holds -e_j*(Attn_ij+1), masked)
  float* evec  = M2f   + 4096;       // 64
  float* bvec  = evec  + 64;         // 64
  float* gvec  = bvec  + 64;         // 64
  float* betav = gvec  + 64;         // 64

  const int tid = threadIdx.x, wave = tid >> 5, lane = tid & 31;
  const int bh = blockIdx.x;
  const int b = bh >> 4, h = bh & 15;
  const long tokBase = (long)b * LSEQ;
  const int colBase = h * HDIM;

  // init per-(b,h) state
  for (int i = tid; i < 4096; i += 128) Wf[i] = W1[h * 4096 + i];
  if (tid < 64) {
    bvec[tid]  = b1[h * 64 + tid];
    gvec[tid]  = tnw[h * 64 + tid];
    betav[tid] = tnb[h * 64 + tid];
  }
  __syncthreads();

  for (int nc = 0; nc < NCHUNK; ++nc) {
    const long t0 = tokBase + (long)nc * CHUNK;
    // ---- load xq, xk chunk tiles + eta vector ----
    for (int i = tid; i < 4096; i += 128) {
      int r = i >> 6, c = i & 63;
      xqf[i] = XQ[(t0 + r) * DIM + colBase + c];
      xkf[i] = XK[(t0 + r) * DIM + colBase + c];
    }
    if (tid < 64) evec[tid] = eta[(long)h * NTOK + t0 + tid];
    __syncthreads();

    // ---- GEMM1: Z1 = xk @ W + b ----
    {
      v8f acc[4] = {};
      for (int kk = 0; kk < 64; kk += 32) {
        v16bf a = load_A_f32(xkf, 64, wave * 16, kk);
#pragma unroll
        for (int t = 0; t < 4; ++t) {
          v16bf bf = load_B_f32(Wf, 64, t * 16, kk);
          acc[t] = wmma_bf16(a, bf, acc[t]);
        }
      }
      store_tile64(Z1f, acc, wave, lane, bvec);
    }
    __syncthreads();

    // ---- grad = ln_l2_bwd(Z1, xv - xk, g, beta), rowwise ----
    if (tid < 64) {
      const int r = tid;
      const float* z = Z1f + r * 64;
      const float* xvg = XV + (t0 + r) * DIM + colBase;
      float s1 = 0.f, s2 = 0.f;
      for (int d = 0; d < 64; ++d) { float v = z[d]; s1 += v; s2 += v * v; }
      float mu = s1 * (1.f / 64.f);
      float var = s2 * (1.f / 64.f) - mu * mu;
      float stdv = sqrtf(var + 1e-6f);
      float inv = 1.f / stdv;
      float sg = 0.f, sgx = 0.f;
      for (int d = 0; d < 64; ++d) {
        float xh = (z[d] - mu) * inv;
        float tgt = xvg[d] - xkf[r * 64 + d];
        float gxh = (gvec[d] * xh + betav[d] - tgt) * gvec[d];
        sg += gxh; sgx += gxh * xh;
      }
      float denom = 1.f / (64.f * stdv);
      for (int d = 0; d < 64; ++d) {
        float xh = (z[d] - mu) * inv;
        float tgt = xvg[d] - xkf[r * 64 + d];
        float gxh = (gvec[d] * xh + betav[d] - tgt) * gvec[d];
        gradf[r * 64 + d] = (64.f * gxh - sg - xh * sgx) * denom;
      }
    }
    __syncthreads();

    // ---- GEMM2: Attn = xq @ xk^T ; M2[i][j] = (j<=i) ? -e[j]*(Attn+1) : 0 ----
    {
      v8f acc[4] = {};
      for (int kk = 0; kk < 64; kk += 32) {
        v16bf a = load_A_f32(xqf, 64, wave * 16, kk);
#pragma unroll
        for (int t = 0; t < 4; ++t) {
          v16bf bf = load_Bt_f32(xkf, 64, t * 16, kk);  // B^T = xk rows
          acc[t] = wmma_bf16(a, bf, acc[t]);
        }
      }
      int hh = lane >> 4;
      int i0 = wave * 16 + 8 * hh;
#pragma unroll
      for (int t = 0; t < 4; ++t) {
        int j = t * 16 + (lane & 15);
        float ej = evec[j];
#pragma unroll
        for (int v = 0; v < 8; ++v) {
          int i = i0 + v;
          M2f[i * 64 + j] = (j <= i) ? (-ej * (acc[t][v] + 1.f)) : 0.f;
        }
      }
    }
    __syncthreads();

    // ---- GEMM3: Z1_bar = xq @ W + M2 @ grad + b ----
    {
      v8f acc[4] = {};
      for (int kk = 0; kk < 64; kk += 32) {
        v16bf a = load_A_f32(xqf, 64, wave * 16, kk);
#pragma unroll
        for (int t = 0; t < 4; ++t) {
          v16bf bf = load_B_f32(Wf, 64, t * 16, kk);
          acc[t] = wmma_bf16(a, bf, acc[t]);
        }
      }
      for (int kk = 0; kk < 64; kk += 32) {
        v16bf a = load_A_f32(M2f, 64, wave * 16, kk);
#pragma unroll
        for (int t = 0; t < 4; ++t) {
          v16bf bf = load_B_f32(gradf, 64, t * 16, kk);
          acc[t] = wmma_bf16(a, bf, acc[t]);
        }
      }
      store_tile64(Z1f, acc, wave, lane, bvec);   // Z1f now holds Z1_bar
    }
    __syncthreads();

    // ---- out = xq + ln_fwd(Z1_bar)  ||  b -= sum_j e_j * grad_j ----
    if (tid < 64) {
      const int r = tid;
      const float* zb = Z1f + r * 64;
      float s1 = 0.f, s2 = 0.f;
      for (int d = 0; d < 64; ++d) { float v = zb[d]; s1 += v; s2 += v * v; }
      float mu = s1 * (1.f / 64.f);
      float var = s2 * (1.f / 64.f) - mu * mu;
      float inv = rsqrtf(var + 1e-6f);
      float* yout = Y + (t0 + r) * DIM + colBase;
      for (int d = 0; d < 64; ++d)
        yout[d] = xqf[r * 64 + d] + gvec[d] * ((zb[d] - mu) * inv) + betav[d];
    } else {
      int n = tid - 64;
      float s = 0.f;
      for (int j = 0; j < 64; ++j) s += evec[j] * gradf[j * 64 + n];
      bvec[n] -= s;
    }
    __syncthreads();

    // ---- GEMM4: W -= (e .* xk)^T @ grad ----
    {
      v8f acc[4] = {};
      for (int kk = 0; kk < 64; kk += 32) {
        v16bf a = load_A_xkeT(xkf, evec, wave * 16, kk);
#pragma unroll
        for (int t = 0; t < 4; ++t) {
          v16bf bf = load_B_f32(gradf, 64, t * 16, kk);
          acc[t] = wmma_bf16(a, bf, acc[t]);
        }
      }
      int hh = lane >> 4;
      int m0w = wave * 16 + 8 * hh;
#pragma unroll
      for (int t = 0; t < 4; ++t) {
        int n = t * 16 + (lane & 15);
#pragma unroll
        for (int v = 0; v < 8; ++v) Wf[(m0w + v) * 64 + n] -= acc[t][v];
      }
    }
    __syncthreads();
  }
}

// ===================== K5: post layernorm -> bf16 =====================
__global__ __launch_bounds__(256) void postnorm_kernel(const float* __restrict__ Yin,
                                                       const float* __restrict__ pw,
                                                       const float* __restrict__ pb,
                                                       unsigned short* __restrict__ Ybf) {
  __shared__ float red1[256], red2[256];
  long tok = blockIdx.x;
  const float* y = Yin + tok * DIM;
  int tid = threadIdx.x;
  float s1 = 0.f, s2 = 0.f;
  for (int i = tid; i < DIM; i += 256) { float v = y[i]; s1 += v; s2 += v * v; }
  red1[tid] = s1; red2[tid] = s2;
  __syncthreads();
  for (int off = 128; off; off >>= 1) {
    if (tid < off) { red1[tid] += red1[tid + off]; red2[tid] += red2[tid + off]; }
    __syncthreads();
  }
  float mu = red1[0] * (1.f / DIM);
  float var = red2[0] * (1.f / DIM) - mu * mu;
  float inv = rsqrtf(var + 1e-6f);
  for (int i = tid; i < DIM; i += 256) {
    float v = pw[i] * ((y[i] - mu) * inv) + pb[i];
    unsigned int u = __float_as_uint(v);
    u = (u + 0x7FFFu + ((u >> 16) & 1u)) >> 16;
    Ybf[tok * DIM + i] = (unsigned short)u;
  }
}

// ===================== Host: kernel_launch =====================
extern "C" void kernel_launch(void* const* d_in, const int* in_sizes, int n_in,
                              void* d_out, int out_size, void* d_ws, size_t ws_size,
                              hipStream_t stream) {
  (void)in_sizes; (void)n_in; (void)out_size; (void)ws_size;
  const float* hs  = (const float*)d_in[0];
  const float* wq  = (const float*)d_in[1];
  const float* wqb = (const float*)d_in[2];
  const float* wk  = (const float*)d_in[3];
  const float* wkb = (const float*)d_in[4];
  const float* wv  = (const float*)d_in[5];
  const float* wvb = (const float*)d_in[6];
  const float* wo  = (const float*)d_in[7];
  const float* wob = (const float*)d_in[8];
  const float* tnw = (const float*)d_in[9];
  const float* tnb = (const float*)d_in[10];
  const float* lrw = (const float*)d_in[11];
  const float* lrb = (const float*)d_in[12];
  const float* W1  = (const float*)d_in[13];
  const float* b1  = (const float*)d_in[14];
  const float* pw  = (const float*)d_in[15];
  const float* pb  = (const float*)d_in[16];

  char* ws = (char*)d_ws;
  size_t off = 0;
  auto alloc = [&](size_t bytes) -> void* {
    void* p = ws + off;
    off += (bytes + 255) & ~(size_t)255;
    return p;
  };
  unsigned short* hs_bf = (unsigned short*)alloc(NTOK * DIM * 2);
  unsigned short* wq_bf = (unsigned short*)alloc((size_t)DIM * DIM * 2);
  unsigned short* wk_bf = (unsigned short*)alloc((size_t)DIM * DIM * 2);
  unsigned short* wv_bf = (unsigned short*)alloc((size_t)DIM * DIM * 2);
  unsigned short* wo_bf = (unsigned short*)alloc((size_t)DIM * DIM * 2);
  float* XQ  = (float*)alloc(NTOK * DIM * 4);
  float* XK  = (float*)alloc(NTOK * DIM * 4);
  float* XV  = (float*)alloc(NTOK * DIM * 4);
  float* ETA = (float*)alloc((size_t)HNUM * NTOK * 4);
  float* Yf  = (float*)alloc(NTOK * DIM * 4);
  unsigned short* yn_bf = hs_bf;  // reuse: hs_bf dead after QKV GEMMs

  // 1) converts
  cvt_f32_bf16<<<4096, 256, 0, stream>>>(hs, hs_bf, NTOK * DIM);
  cvt_f32_bf16<<<1024, 256, 0, stream>>>(wq, wq_bf, (long)DIM * DIM);
  cvt_f32_bf16<<<1024, 256, 0, stream>>>(wk, wk_bf, (long)DIM * DIM);
  cvt_f32_bf16<<<1024, 256, 0, stream>>>(wv, wv_bf, (long)DIM * DIM);
  cvt_f32_bf16<<<1024, 256, 0, stream>>>(wo, wo_bf, (long)DIM * DIM);

  // 2) Q/K/V projections (hs @ W^T + b): 128x128 tiles, async double-buffered, bf16 WMMA
  dim3 gemmGrid(DIM / 128, NTOK / 128);  // (8, 256)
  size_t gemmLds = 65536;                // 2 x (16KB A + 16KB B)
  gemm_bf16_wmma<<<gemmGrid, 256, gemmLds, stream>>>(hs_bf, wq_bf, wqb, XQ, (int)NTOK, DIM, DIM);
  gemm_bf16_wmma<<<gemmGrid, 256, gemmLds, stream>>>(hs_bf, wk_bf, wkb, XK, (int)NTOK, DIM, DIM);
  gemm_bf16_wmma<<<gemmGrid, 256, gemmLds, stream>>>(hs_bf, wv_bf, wvb, XV, (int)NTOK, DIM, DIM);

  // 3) eta (learning-rate gate), 4) pre-norms
  eta_kernel<<<(unsigned)NTOK, 256, 0, stream>>>(hs, lrw, lrb, ETA);
  prep_kernel<<<(unsigned)((NTOK * HNUM + 255) / 256), 256, 0, stream>>>(XQ, XK, XV, tnw, tnb);

  // 5) sequential TTT scan: one block per (b,h), fp32 state in LDS, bf16 WMMA tiles
  size_t scanLds = (6 * 4096 + 4 * 64) * sizeof(float);  // ~97 KB (CDNA5 LDS: 320 KB/WG)
  ttt_scan<<<BATCH * HNUM, 128, scanLds, stream>>>(XQ, XK, XV, ETA, tnw, tnb, W1, b1, Yf);

  // 6) post layernorm -> bf16, 7) output projection
  postnorm_kernel<<<(unsigned)NTOK, 256, 0, stream>>>(Yf, pw, pb, yn_bf);
  gemm_bf16_wmma<<<gemmGrid, 256, gemmLds, stream>>>(yn_bf, wo_bf, wob, (float*)d_out, (int)NTOK, DIM, DIM);
}
